// SemiSupervisedNetwork_37915971289209
// MI455X (gfx1250) — matrix operand order repaired and checked by reading, not compiled, not run on hardware
//
#include <hip/hip_runtime.h>

typedef _Float16 half_t;
typedef __attribute__((ext_vector_type(16))) _Float16 v16h;
typedef __attribute__((ext_vector_type(8)))  _Float16 v8h;
typedef __attribute__((ext_vector_type(8)))  float    v8f;

// Model dims
#define BB    128
#define NIN   784
#define NINP  832      // K padded to multiple of 64
#define NHID  2048
#define NOUT  512
#define TT    250
#define MM    (TT * BB)   // 32000 rows for both big GEMMs

// LDS row stride for staged tiles: 64 K-halves + 8 pad = 72 halves (144 B,
// 16B-aligned for b128, bank-conflict-free fragment reads)
#define LSTR  72

// ---------------------------------------------------------------------------
// LDS helpers: byte offset of a shared-memory pointer (for async-to-LDS asm)
// ---------------------------------------------------------------------------
typedef const __attribute__((address_space(3))) void* lds_cptr;
__device__ __forceinline__ unsigned lds_off(const void* p) {
  return (unsigned)(unsigned long long)(lds_cptr)p;
}
__device__ __forceinline__ void async_copy_b128(unsigned lds_byte_off, const void* g) {
  unsigned long long ga = (unsigned long long)(uintptr_t)g;
  asm volatile("global_load_async_to_lds_b128 %0, %1, off"
               :: "v"(lds_byte_off), "v"(ga) : "memory");
}
__device__ __forceinline__ void wait_async0() {
  asm volatile("s_wait_asynccnt 0" ::: "memory");
}

// ---------------------------------------------------------------------------
// ReLU + f32->f16 + transpose: W (Kreal x N) -> Wt (N x Kpad), zero K-padding.
// ---------------------------------------------------------------------------
__global__ void relu_cvt_transpose(const float* __restrict__ w, half_t* __restrict__ wt,
                                   int Kreal, int Kpad, int N) {
  __shared__ float tile[32][33];
  const int k0 = blockIdx.x * 32, n0 = blockIdx.y * 32;
  const int tx = threadIdx.x, ty = threadIdx.y;
  float v = 0.0f;
  if (k0 + ty < Kreal) v = fmaxf(w[(long)(k0 + ty) * N + n0 + tx], 0.0f);
  tile[ty][tx] = v;
  __syncthreads();
  const int nn = n0 + ty, kk = k0 + tx;
  if (kk < Kpad) wt[(long)nn * Kpad + kk] = (half_t)tile[tx][ty];
}

// ---------------------------------------------------------------------------
// Pack input spikes (b, i, t) f32 -> A1[(t*B + b)*NINP + i] f16 (zero-padded)
// ---------------------------------------------------------------------------
__global__ void pack_spikes(const float* __restrict__ in, half_t* __restrict__ out) {
  __shared__ float tile[32][33];
  const int b  = blockIdx.z;
  const int i0 = blockIdx.x * 32, t0 = blockIdx.y * 32;
  const int tx = threadIdx.x, ty = threadIdx.y;
  float v = 0.0f;
  if (i0 + ty < NIN && t0 + tx < TT) v = in[((long)b * NIN + i0 + ty) * TT + t0 + tx];
  tile[ty][tx] = v;
  __syncthreads();
  const int tt = t0 + ty, ii = i0 + tx;
  if (tt < TT && ii < NINP)
    out[((long)tt * BB + b) * NINP + ii] = (half_t)tile[tx][ty];
}

// ---------------------------------------------------------------------------
// f16 x f16 -> f32 WMMA GEMM.  C[M x N] = A[M x K] * Bt[N x K]^T
// Block tile 128x128, 8 waves (4 along M x 2 along N), wave tile 32x64.
// K chunk 64 (two 16x16x32 sub-chunks), double-buffered async-to-LDS staging,
// one barrier per chunk. N, K compile-time; M%128==0, N%128==0, K%64==0.
// ---------------------------------------------------------------------------
template <int N, int K>
__global__ __launch_bounds__(256) void wmma_gemm_f16(
    const half_t* __restrict__ A, const half_t* __restrict__ Bt,
    float* __restrict__ C) {
  __shared__ __align__(16) half_t AsBuf[2][128 * LSTR];
  __shared__ __align__(16) half_t BsBuf[2][128 * LSTR];

  const int tid  = threadIdx.x;
  const int wid  = tid >> 5;
  const int lane = tid & 31;
  const int m0 = blockIdx.y * 128;
  const int n0 = blockIdx.x * 128;
  const int waveM = (wid >> 1) * 32;   // 4 waves along M
  const int waveN = (wid & 1) * 64;    // 2 waves along N

  const unsigned asOff[2] = { lds_off(&AsBuf[0][0]), lds_off(&AsBuf[1][0]) };
  const unsigned bsOff[2] = { lds_off(&BsBuf[0][0]), lds_off(&BsBuf[1][0]) };

  const half_t* Ab = A  + (long)m0 * K;   // 128 x K panel
  const half_t* Bb = Bt + (long)n0 * K;   // 128 x K panel

  // stage one 128x64-half tile: 1024 x 16B DMA transfers, 4 per thread.
  auto stage = [&](const half_t* gpanel, unsigned ldsbase, int kc) {
#pragma unroll
    for (int j = 0; j < 4; ++j) {
      int sid = tid + 256 * j;          // 0..1023
      int row = sid >> 3, seg = sid & 7;
      async_copy_b128(ldsbase + (unsigned)(row * (LSTR * 2) + seg * 16),
                      gpanel + (long)row * K + kc + seg * 8);
    }
  };

  v8f acc[2][4] = {};

  const int frow  = lane & 15;
  const int khalf = lane >> 4;
  constexpr int NCH = K / 64;

  // prologue: chunk 0 -> buffer 0
  stage(Ab, asOff[0], 0);
  stage(Bb, bsOff[0], 0);

  for (int c = 0; c < NCH; ++c) {
    wait_async0();        // chunk c DMA complete (this thread's issues)
    __syncthreads();      // lowers to s_wait_dscnt 0 + barrier: tile resident
                          // AND all prior reads of the other buffer retired
    const int cur = c & 1;
    if (c + 1 < NCH) {    // overlap next chunk's DMA with this chunk's math
      stage(Ab, asOff[cur ^ 1], (c + 1) * 64);
      stage(Bb, bsOff[cur ^ 1], (c + 1) * 64);
    }

#pragma unroll
    for (int ks = 0; ks < 2; ++ks) {    // two 16x16x32 sub-chunks of K
      v16h afrag[2], bfrag[4];
#pragma unroll
      for (int sm = 0; sm < 2; ++sm) {
        const half_t* base =
            &AsBuf[cur][(waveM + sm * 16 + frow) * LSTR + ks * 32 + khalf * 8];
        v8h lo = *(const v8h*)(base);        // K 0..7   (lanes 16-31: 8..15)
        v8h hi = *(const v8h*)(base + 16);   // K 16..23 (lanes 16-31: 24..31)
#pragma unroll
        for (int i = 0; i < 8; ++i) { afrag[sm][i] = lo[i]; afrag[sm][i + 8] = hi[i]; }
      }
#pragma unroll
      for (int sn = 0; sn < 4; ++sn) {
        const half_t* base =
            &BsBuf[cur][(waveN + sn * 16 + frow) * LSTR + ks * 32 + khalf * 8];
        v8h lo = *(const v8h*)(base);        // K 0..7   (lanes 16-31: 8..15)
        v8h hi = *(const v8h*)(base + 16);   // K 16..23 (lanes 16-31: 24..31)
#pragma unroll
        for (int i = 0; i < 8; ++i) { bfrag[sn][i] = lo[i]; bfrag[sn][i + 8] = hi[i]; }
      }
#pragma unroll
      for (int sm = 0; sm < 2; ++sm)
#pragma unroll
        for (int sn = 0; sn < 4; ++sn)
          acc[sm][sn] = __builtin_amdgcn_wmma_f32_16x16x32_f16(
              false, afrag[sm], false, bfrag[sn], (short)0, acc[sm][sn], false, false);
    }
    // no trailing barrier: next iteration's wait+__syncthreads provides the
    // read-before-overwrite guarantee.
  }

  // ---- epilogue: C/D layout row = r + 8*(lane>>4), col = lane&15 ----------
  const int lh = lane >> 4, lc = lane & 15;
  float* cbase = C + (long)(m0 + waveM + 8 * lh) * N + (n0 + waveN + lc);
#pragma unroll
  for (int sm = 0; sm < 2; ++sm)
#pragma unroll
    for (int sn = 0; sn < 4; ++sn)
#pragma unroll
      for (int r = 0; r < 8; ++r)
        cbase[(long)(sm * 16 + r) * N + sn * 16] = acc[sm][sn][r];
}

// ---------------------------------------------------------------------------
// Hidden LIF scan: one thread per (b,h), 250 in-register steps. Emits
// hidden_spikes (b,h,t) f32 and the time-shifted fp16 spike matrix S
// (row t+1 <- spikes(t); row block 0 zeroed) feeding GEMM2 directly.
// ---------------------------------------------------------------------------
__global__ void lif_hidden(const float* __restrict__ Ih, float* __restrict__ out_h,
                           half_t* __restrict__ S) {
  const int idx = blockIdx.x * 256 + threadIdx.x;     // b*NHID + h
  S[idx] = (half_t)0.0f;                              // zero t=0 rows of S
  float v = 0.0f;
  float2 buf;
  const long obase = (long)idx * TT;
  const long step = (long)BB * NHID;
  for (int t = 0; t < TT; ++t) {
    float I = Ih[(long)t * step + idx];
    float vn = v + 0.1f * (0.0f - v) + I;             // DT/TAU=0.1, V_REST=0
    float s = (vn >= 1.0f) ? 1.0f : 0.0f;
    v = (s > 0.0f) ? 0.0f : vn;                       // hard reset
    if ((t & 1) == 0) buf.x = s;
    else { buf.y = s; *(float2*)(out_h + obase + t - 1) = buf; }
    if (t < TT - 1) S[(long)(t + 1) * step + idx] = (half_t)s;
  }
}

// ---------------------------------------------------------------------------
// Output LIF scan: one thread per (b,o); also accumulates firing rates.
// ---------------------------------------------------------------------------
__global__ void lif_output(const float* __restrict__ Io, float* __restrict__ out_o,
                           float* __restrict__ rates) {
  const int idx = blockIdx.x * 256 + threadIdx.x;     // b*NOUT + o
  float v = 0.0f, sum = 0.0f;
  float2 buf;
  const long obase = (long)idx * TT;
  const long step = (long)BB * NOUT;
  for (int t = 0; t < TT; ++t) {
    float I = Io[(long)t * step + idx];               // I_o(t) = s_h(t-1) . W
    float vn = v + 0.1f * (0.0f - v) + I;
    float s = (vn >= 1.0f) ? 1.0f : 0.0f;
    v = (s > 0.0f) ? 0.0f : vn;
    sum += s;
    if ((t & 1) == 0) buf.x = s;
    else { buf.y = s; *(float2*)(out_o + obase + t - 1) = buf; }
  }
  rates[idx] = sum * (1.0f / (float)TT);
}

// ---------------------------------------------------------------------------
extern "C" void kernel_launch(void* const* d_in, const int* in_sizes, int n_in,
                              void* d_out, int out_size, void* d_ws, size_t ws_size,
                              hipStream_t stream) {
  (void)in_sizes; (void)n_in; (void)out_size; (void)ws_size;
  const float* in_spikes = (const float*)d_in[0];   // (B, NIN, T) f32
  const float* w_ih      = (const float*)d_in[1];   // (NIN, NHID) f32
  const float* w_ho      = (const float*)d_in[2];   // (NHID, NOUT) f32

  float* out        = (float*)d_out;
  float* out_hidden = out;                                   // B*NHID*T
  float* out_output = out_hidden + (long)BB * NHID * TT;     // B*NOUT*T
  float* out_rates  = out_output + (long)BB * NOUT * TT;     // B*NOUT

  // Workspace carve-up (all chunks multiples of 256 bytes)
  char* ws = (char*)d_ws;
  half_t* A1   = (half_t*)ws; ws += (long)MM * NINP * 2;     //  53.2 MB spikes f16
  half_t* WihT = (half_t*)ws; ws += (long)NHID * NINP * 2;   //   3.4 MB (NHID x NINP)
  half_t* WhoT = (half_t*)ws; ws += (long)NOUT * NHID * 2;   //   2.1 MB (NOUT x NHID)
  float*  Ih   = (float*) ws; ws += (long)MM * NHID * 4;     // 262.1 MB hidden currents
  half_t* S    = (half_t*)ws; ws += (long)MM * NHID * 2;     // 131.1 MB shifted spikes
  float*  Io   = (float*) ws; ws += (long)MM * NOUT * 4;     //  65.5 MB output currents

  // 1) weight conversion: relu + f16 + transpose (+ K zero-padding for Wih)
  relu_cvt_transpose<<<dim3(NINP / 32, NHID / 32), dim3(32, 32), 0, stream>>>(
      w_ih, WihT, NIN, NINP, NHID);
  relu_cvt_transpose<<<dim3(NHID / 32, NOUT / 32), dim3(32, 32), 0, stream>>>(
      w_ho, WhoT, NHID, NHID, NOUT);

  // 2) pack + transpose spikes to f16 (t*B+b, i_pad)
  pack_spikes<<<dim3(NINP / 32, (TT + 31) / 32, BB), dim3(32, 32), 0, stream>>>(
      in_spikes, A1);

  // 3) GEMM1: all hidden input currents. M=32000, N=2048, K=832
  wmma_gemm_f16<NHID, NINP><<<dim3(NHID / 128, MM / 128), 256, 0, stream>>>(
      A1, WihT, Ih);

  // 4) hidden LIF scan -> hidden_spikes + shifted f16 spike matrix S
  lif_hidden<<<(BB * NHID) / 256, 256, 0, stream>>>(Ih, out_hidden, S);

  // 5) GEMM2: all output currents from previous-step spikes. M=32000, N=512, K=2048
  wmma_gemm_f16<NOUT, NHID><<<dim3(NOUT / 128, MM / 128), 256, 0, stream>>>(
      S, WhoT, Io);

  // 6) output LIF scan -> output_spikes + firing rates
  lif_output<<<(BB * NOUT) / 256, 256, 0, stream>>>(Io, out_output, out_rates);
}
